// TFAttention_25443386262197
// MI455X (gfx1250) — compile-verified, matrix operand order checked by reading, not run on hardware
//
#include <hip/hip_runtime.h>

// Problem dims
#define BQ   4
#define SS   2048
#define NXX  1024
#define NHH  16
#define HDD  64
#define MTOT (BQ * SS)            // 8192 rows for the two big GEMMs
#define AELEMS ((size_t)BQ * SS * NXX)  // 8388608: size of 'a' and of q/k/v each

typedef __attribute__((ext_vector_type(16))) __bf16 v16bf;
typedef __attribute__((ext_vector_type(8)))  __bf16 bf16x8;
typedef __attribute__((ext_vector_type(4)))  __bf16 bf16x4;
typedef __attribute__((ext_vector_type(2)))  __bf16 bf16x2;
typedef __attribute__((ext_vector_type(8)))  float  v8f;
typedef __attribute__((ext_vector_type(4)))  float  f32x4;
typedef __attribute__((ext_vector_type(4)))  unsigned int u32x4;
typedef __attribute__((ext_vector_type(8)))  int    i32x8;
typedef __attribute__((ext_vector_type(4)))  int    i32x4;

#if __has_builtin(__builtin_amdgcn_tensor_load_to_lds) && __has_builtin(__builtin_amdgcn_s_wait_tensorcnt)
#define HAVE_TDM 1
#else
#define HAVE_TDM 0
#endif

union FragAB { v16bf v; bf16x8 h[2]; };

__device__ inline __bf16 f2bf(float f) { return (__bf16)f; }

// pack two floats -> one dword of two bf16 (lowers to v_cvt_pk_bf16_f32)
__device__ inline bf16x2 pack2(float lo, float hi) {
    bf16x2 r; r[0] = f2bf(lo); r[1] = f2bf(hi); return r;
}

__device__ inline v8f wmma_bf16(v16bf a, v16bf b, v8f c) {
    // D = A(16x32 bf16) x B(32x16 bf16) + C(16x16 f32)
    return __builtin_amdgcn_wmma_f32_16x16x32_bf16(
        /*neg_a=*/false, a, /*neg_b=*/false, b,
        /*c_mod=*/(short)0, c, /*reuse_a=*/false, /*reuse_b=*/false);
}

// ---------------------------------------------------------------------------
// Kernel 1: qkv = x @ c_attn_w + b ; scatter K,V (f32) into `present`,
//           Q (bf16, [B,NH,S,HD]) into workspace.
// Block tile 128(M) x 64(N), 8 waves of 32x32, K-step 32.
// ---------------------------------------------------------------------------
#define LDA 40   // padded LDS row stride for 32-element rows (80B, 16B-aligned)

__global__ __launch_bounds__(256) void qkv_gemm_kernel(
    const float* __restrict__ x, const float* __restrict__ w,
    const float* __restrict__ bias, float* __restrict__ present,
    __bf16* __restrict__ qws)
{
    __shared__ __align__(16) __bf16 sA[128 * LDA];
    __shared__ __align__(16) __bf16 sB[64 * LDA];

    const int tid   = threadIdx.x;
    const int lane  = tid & 31;
    const int wave  = tid >> 5;
    const int waveM = wave >> 1;    // 0..3
    const int waveN = wave & 1;     // 0..1
    const int rowSel = lane & 15;
    const int kh     = lane >> 4;   // K-half select per WMMA lane layout
    const int blockN = blockIdx.x * 64;
    const int blockM = blockIdx.y * 128;

    v8f zero8 = {0.f,0.f,0.f,0.f,0.f,0.f,0.f,0.f};
    v8f acc[2][2];
    acc[0][0] = zero8; acc[0][1] = zero8; acc[1][0] = zero8; acc[1][1] = zero8;

    for (int kb = 0; kb < NXX / 32; ++kb) {
        // Stage A (x, fp32 -> bf16), 128x32
        #pragma unroll
        for (int r = 0; r < 4; ++r) {
            int id  = tid + 256 * r;
            int row = id >> 3, f4 = id & 7;
            f32x4 v = *(const f32x4*)(x + (size_t)(blockM + row) * NXX + kb * 32 + f4 * 4);
            bf16x4 hv;
            hv[0] = f2bf(v[0]); hv[1] = f2bf(v[1]); hv[2] = f2bf(v[2]); hv[3] = f2bf(v[3]);
            *(bf16x4*)&sA[row * LDA + f4 * 4] = hv;
        }
        // Stage B (W, fp32 -> bf16, transposed to [n][k]), 32x64:
        // each thread owns a 2(k)x4(n) sub-block; packed b32 transposed stores
        {
            int k  = (tid >> 4) * 2;          // 0..30
            int n4 = (tid & 15) * 4;          // 0..60
            const float* wp = w + (size_t)(kb * 32 + k) * (3 * NXX) + blockN + n4;
            f32x4 r0 = *(const f32x4*)(wp);
            f32x4 r1 = *(const f32x4*)(wp + 3 * NXX);
            #pragma unroll
            for (int u = 0; u < 4; ++u)
                *(bf16x2*)&sB[(n4 + u) * LDA + k] = pack2(r0[u], r1[u]);
        }
        __syncthreads();

        if (kb + 1 < NXX / 32) {   // global_prefetch_b8 of next A tile
            __builtin_prefetch(x + (size_t)(blockM + (tid >> 3)) * NXX + (kb + 1) * 32, 0, 1);
        }

        FragAB aF[2], bF[2];
        #pragma unroll
        for (int tm = 0; tm < 2; ++tm) {
            int row = waveM * 32 + tm * 16 + rowSel;
            aF[tm].h[0] = *(const bf16x8*)&sA[row * LDA + kh * 8];
            aF[tm].h[1] = *(const bf16x8*)&sA[row * LDA + 16 + kh * 8];
        }
        #pragma unroll
        for (int tn = 0; tn < 2; ++tn) {
            int col = waveN * 32 + tn * 16 + rowSel;
            bF[tn].h[0] = *(const bf16x8*)&sB[col * LDA + kh * 8];
            bF[tn].h[1] = *(const bf16x8*)&sB[col * LDA + 16 + kh * 8];
        }
        #pragma unroll
        for (int tm = 0; tm < 2; ++tm)
            #pragma unroll
            for (int tn = 0; tn < 2; ++tn)
                acc[tm][tn] = wmma_bf16(aF[tm].v, bF[tn].v, acc[tm][tn]);
        __syncthreads();
    }

    // Epilogue: route q -> ws(bf16), k/v -> present(f32)
    #pragma unroll
    for (int tm = 0; tm < 2; ++tm) {
        #pragma unroll
        for (int tn = 0; tn < 2; ++tn) {
            #pragma unroll
            for (int j = 0; j < 8; ++j) {
                int m = blockM + waveM * 32 + tm * 16 + j + 8 * kh;
                int n = blockN + waveN * 32 + tn * 16 + rowSel;
                float val = acc[tm][tn][j] + bias[n];
                int b = m >> 11, s = m & (SS - 1);
                int which = n >> 10, c = n & (NXX - 1);
                int hh = c >> 6, d = c & 63;
                if (which == 0) {
                    qws[(((size_t)(b * NHH + hh)) * SS + s) * HDD + d] = f2bf(val);
                } else {
                    present[((((size_t)(b * 2 + (which - 1))) * NHH + hh) * SS + s) * HDD + d] = val;
                }
            }
        }
    }
}

// ---------------------------------------------------------------------------
// Kernel 2: causal flash attention per (b, h, 64 q-rows). 4 waves, each owning
// a 16-row q tile; kv processed in chunks of 32 with online softmax; output
// written merged-head bf16 [B,S,NX] into workspace.
// ---------------------------------------------------------------------------
#define LDK 72   // sK row stride (64 hd elements + pad), 144B, 16B-aligned
#define LDV 40   // sV row stride (32 kv elements + pad)
#define LDP 40   // sP row stride (32 kv elements + pad)

__global__ __launch_bounds__(128) void attn_flash_kernel(
    const __bf16* __restrict__ qws, const float* __restrict__ present,
    __bf16* __restrict__ aws)
{
    __shared__ __align__(16) __bf16 sK[32 * LDK];      // [kv][hd]
    __shared__ __align__(16) __bf16 sV[64 * LDV];      // [hd][kv] (transposed)
    __shared__ __align__(16) __bf16 sP[4][16 * LDP];   // per-wave P tile

    const int tid  = threadIdx.x;
    const int lane = tid & 31;
    const int wave = tid >> 5;
    const int rowSel = lane & 15;
    const int kh     = lane >> 4;
    const int qt = blockIdx.x, h = blockIdx.y, b = blockIdx.z;
    const int qbase = qt * 64 + wave * 16;

    const __bf16* qptr = qws + ((size_t)(b * NHH + h)) * SS * HDD;
    const float*  kptr = present + (((size_t)(b * 2 + 0)) * NHH + h) * SS * HDD;
    const float*  vptr = present + (((size_t)(b * 2 + 1)) * NHH + h) * SS * HDD;

    // Q fragments (K-dim = 64 -> two WMMA k-steps), kept in VGPRs all loop
    FragAB qF[2];
    #pragma unroll
    for (int ks = 0; ks < 2; ++ks) {
        const __bf16* rp = qptr + (size_t)(qbase + rowSel) * HDD + ks * 32;
        qF[ks].h[0] = *(const bf16x8*)(rp + kh * 8);
        qF[ks].h[1] = *(const bf16x8*)(rp + 16 + kh * 8);
    }

    v8f zero8 = {0.f,0.f,0.f,0.f,0.f,0.f,0.f,0.f};
    v8f O[4]; O[0] = zero8; O[1] = zero8; O[2] = zero8; O[3] = zero8;
    float mrow[8], lrow[8];
    #pragma unroll
    for (int j = 0; j < 8; ++j) { mrow[j] = -1e30f; lrow[j] = 0.f; }

    const int nT = 2 * qt + 2;  // causal: kv tiles up to block diagonal
    for (int t = 0; t < nT; ++t) {
        const int kvbase = t * 32;
        // Stage K -> sK[kv][hd] (row-major bf16x4 stores) and
        //       V -> sV[hd][kv] (packed b32 transposed stores).
        // Each thread owns a 2(kv)x4(hd) sub-block, 2 blocks per thread.
        #pragma unroll
        for (int r = 0; r < 2; ++r) {
            int id  = tid + 128 * r;
            int kv  = (id >> 4) * 2;      // 0..30
            int d4  = (id & 15) * 4;      // 0..60
            const float* kp = kptr + (size_t)(kvbase + kv) * HDD + d4;
            f32x4 k0 = *(const f32x4*)(kp);
            f32x4 k1 = *(const f32x4*)(kp + HDD);
            bf16x4 hk0, hk1;
            hk0[0]=f2bf(k0[0]); hk0[1]=f2bf(k0[1]); hk0[2]=f2bf(k0[2]); hk0[3]=f2bf(k0[3]);
            hk1[0]=f2bf(k1[0]); hk1[1]=f2bf(k1[1]); hk1[2]=f2bf(k1[2]); hk1[3]=f2bf(k1[3]);
            *(bf16x4*)&sK[kv * LDK + d4]       = hk0;
            *(bf16x4*)&sK[(kv + 1) * LDK + d4] = hk1;
            const float* vp = vptr + (size_t)(kvbase + kv) * HDD + d4;
            f32x4 v0 = *(const f32x4*)(vp);
            f32x4 v1 = *(const f32x4*)(vp + HDD);
            #pragma unroll
            for (int u = 0; u < 4; ++u)
                *(bf16x2*)&sV[(d4 + u) * LDV + kv] = pack2(v0[u], v1[u]);
        }
        __syncthreads();

        // Scores: two 16x16 tiles (kv cols 0..15, 16..31), K-dim 64 over hd
        v8f sc[2]; sc[0] = zero8; sc[1] = zero8;
        #pragma unroll
        for (int nt = 0; nt < 2; ++nt) {
            int col = nt * 16 + rowSel;
            #pragma unroll
            for (int ks = 0; ks < 2; ++ks) {
                FragAB kF;
                kF.h[0] = *(const bf16x8*)&sK[col * LDK + ks * 32 + kh * 8];
                kF.h[1] = *(const bf16x8*)&sK[col * LDK + ks * 32 + 16 + kh * 8];
                sc[nt] = wmma_bf16(qF[ks].v, kF.v, sc[nt]);
            }
        }

        // Online softmax (rows live as: VGPR j -> row j + 8*kh, col = rowSel)
        __bf16* sPw = &sP[wave][0];
        #pragma unroll
        for (int j = 0; j < 8; ++j) {
            int qr  = qbase + j + 8 * kh;
            int kc0 = kvbase + rowSel;
            int kc1 = kvbase + 16 + rowSel;
            float s0 = (kc0 <= qr) ? sc[0][j] * 0.125f : -10000.0f;
            float s1 = (kc1 <= qr) ? sc[1][j] * 0.125f : -10000.0f;
            float tmax = fmaxf(s0, s1);
            #pragma unroll
            for (int msk = 1; msk < 16; msk <<= 1)
                tmax = fmaxf(tmax, __shfl_xor(tmax, msk, 32));
            float mn = fmaxf(mrow[j], tmax);
            float scale = __expf(mrow[j] - mn);
            float p0 = __expf(s0 - mn);
            float p1 = __expf(s1 - mn);
            float rs = p0 + p1;
            #pragma unroll
            for (int msk = 1; msk < 16; msk <<= 1)
                rs += __shfl_xor(rs, msk, 32);
            lrow[j] = lrow[j] * scale + rs;
            mrow[j] = mn;
            #pragma unroll
            for (int dt = 0; dt < 4; ++dt) O[dt][j] *= scale;
            int prow = j + 8 * kh;
            sPw[prow * LDP + rowSel]      = f2bf(p0);
            sPw[prow * LDP + 16 + rowSel] = f2bf(p1);
        }

        // PV: P(16x32) x V(32x64) accumulated into O (4 d-tiles)
        FragAB pF;
        pF.h[0] = *(const bf16x8*)&sPw[rowSel * LDP + kh * 8];
        pF.h[1] = *(const bf16x8*)&sPw[rowSel * LDP + 16 + kh * 8];
        #pragma unroll
        for (int dt = 0; dt < 4; ++dt) {
            int col = dt * 16 + rowSel;
            FragAB vF;
            vF.h[0] = *(const bf16x8*)&sV[col * LDV + kh * 8];
            vF.h[1] = *(const bf16x8*)&sV[col * LDV + 16 + kh * 8];
            O[dt] = wmma_bf16(pF.v, vF.v, O[dt]);
        }
        __syncthreads();
    }

    // Normalize and store merged-head bf16 [B,S,NX]
    #pragma unroll
    for (int j = 0; j < 8; ++j) {
        float inv = 1.0f / lrow[j];
        int srow = qbase + j + 8 * kh;
        size_t base = ((size_t)b * SS + srow) * NXX + h * HDD;
        #pragma unroll
        for (int dt = 0; dt < 4; ++dt)
            aws[base + dt * 16 + rowSel] = f2bf(O[dt][j] * inv);
    }
}

// ---------------------------------------------------------------------------
// Kernel 3: a = attn @ c_proj_w + b  (A already bf16 in ws)
// A tile (128x32 bf16) is staged by the Tensor Data Mover: one wave issues a
// 2D tensor_load_to_lds with LDS padding fields reproducing the LDA=40 stride.
// ---------------------------------------------------------------------------
__global__ __launch_bounds__(256) void proj_gemm_kernel(
    const __bf16* __restrict__ aws, const float* __restrict__ w,
    const float* __restrict__ bias, float* __restrict__ out)
{
    __shared__ __align__(16) __bf16 sA[128 * LDA];
    __shared__ __align__(16) __bf16 sB[64 * LDA];

    const int tid   = threadIdx.x;
    const int lane  = tid & 31;
    const int wave  = tid >> 5;
    const int waveM = wave >> 1;
    const int waveN = wave & 1;
    const int rowSel = lane & 15;
    const int kh     = lane >> 4;
    const int blockN = blockIdx.x * 64;
    const int blockM = blockIdx.y * 128;

    v8f zero8 = {0.f,0.f,0.f,0.f,0.f,0.f,0.f,0.f};
    v8f acc[2][2];
    acc[0][0] = zero8; acc[0][1] = zero8; acc[1][0] = zero8; acc[1][1] = zero8;

    for (int kb = 0; kb < NXX / 32; ++kb) {
#if HAVE_TDM
        // --- TDM path: DMA the 128x32 bf16 A tile into LDS with row padding.
        if (wave == 0) {
            unsigned long long ga =
                (unsigned long long)(const void*)(aws + (size_t)blockM * NXX + kb * 32);
            unsigned int ldsa = (unsigned int)(unsigned long long)(const void*)&sA[0];
            u32x4 g0;
            g0[0] = 1u;                                   // count=1, user mode
            g0[1] = ldsa;                                 // lds_addr
            g0[2] = (unsigned int)(ga & 0xFFFFFFFFu);     // global_addr lo
            g0[3] = (unsigned int)((ga >> 32) & 0x1FFFFFFu) | (2u << 30); // addr hi | type=2
            i32x8 g1;
            // data_size=1 (2B), pad_enable, pad_interval=3 (16 DW = 64B row),
            // pad_amount=3 (4 DW = 16B) -> LDS row stride 80B == LDA
            g1[0] = (int)((1u << 16) | (1u << 20) | (3u << 22) | (3u << 25));
            g1[1] = (int)((NXX & 0xFFFFu) << 16);                       // tensor_dim0 lo
            g1[2] = (int)((NXX >> 16) | ((MTOT & 0xFFFFu) << 16));      // td0 hi | td1 lo
            g1[3] = (int)((MTOT >> 16) | (32u << 16));                  // td1 hi | tile_dim0=32
            g1[4] = (int)(128u);                                        // tile_dim1=128
            g1[5] = (int)(NXX);                                         // tensor_dim0_stride lo
            g1[6] = 0;                                                  // stride0 hi | stride1 lo
            g1[7] = 0;
            i32x4 gz4 = {0, 0, 0, 0};
            i32x8 gz8 = {0, 0, 0, 0, 0, 0, 0, 0};
            __builtin_amdgcn_tensor_load_to_lds(g0, g1, gz4, gz4, gz8, 0);
            __builtin_amdgcn_s_wait_tensorcnt(0);
        }
#else
        // --- Fallback: vector-load A tile (already bf16): 128x32, 16B chunks
        #pragma unroll
        for (int r = 0; r < 2; ++r) {
            int id = tid + 256 * r;
            int row = id >> 2, c8 = id & 3;
            *(bf16x8*)&sA[row * LDA + c8 * 8] =
                *(const bf16x8*)(aws + (size_t)(blockM + row) * NXX + kb * 32 + c8 * 8);
        }
#endif
        // Stage B (W f32 -> bf16, transposed): 2(k)x4(n) sub-block per thread
        {
            int k  = (tid >> 4) * 2;
            int n4 = (tid & 15) * 4;
            const float* wp = w + (size_t)(kb * 32 + k) * NXX + blockN + n4;
            f32x4 r0 = *(const f32x4*)(wp);
            f32x4 r1 = *(const f32x4*)(wp + NXX);
            #pragma unroll
            for (int u = 0; u < 4; ++u)
                *(bf16x2*)&sB[(n4 + u) * LDA + k] = pack2(r0[u], r1[u]);
        }
        __syncthreads();

        if (kb + 1 < NXX / 32) {   // prefetch next W tile
            __builtin_prefetch(w + (size_t)((kb + 1) * 32 + (tid >> 4) * 2) * NXX + blockN, 0, 1);
        }

        FragAB aF[2], bF[2];
        #pragma unroll
        for (int tm = 0; tm < 2; ++tm) {
            int row = waveM * 32 + tm * 16 + rowSel;
            aF[tm].h[0] = *(const bf16x8*)&sA[row * LDA + kh * 8];
            aF[tm].h[1] = *(const bf16x8*)&sA[row * LDA + 16 + kh * 8];
        }
        #pragma unroll
        for (int tn = 0; tn < 2; ++tn) {
            int col = waveN * 32 + tn * 16 + rowSel;
            bF[tn].h[0] = *(const bf16x8*)&sB[col * LDA + kh * 8];
            bF[tn].h[1] = *(const bf16x8*)&sB[col * LDA + 16 + kh * 8];
        }
        #pragma unroll
        for (int tm = 0; tm < 2; ++tm)
            #pragma unroll
            for (int tn = 0; tn < 2; ++tn)
                acc[tm][tn] = wmma_bf16(aF[tm].v, bF[tn].v, acc[tm][tn]);
        __syncthreads();
    }

    #pragma unroll
    for (int tm = 0; tm < 2; ++tm) {
        #pragma unroll
        for (int tn = 0; tn < 2; ++tn) {
            #pragma unroll
            for (int j = 0; j < 8; ++j) {
                int m = blockM + waveM * 32 + tm * 16 + j + 8 * kh;
                int n = blockN + waveN * 32 + tn * 16 + rowSel;
                out[(size_t)m * NXX + n] = acc[tm][tn][j] + bias[n];
            }
        }
    }
}

// ---------------------------------------------------------------------------
extern "C" void kernel_launch(void* const* d_in, const int* in_sizes, int n_in,
                              void* d_out, int out_size, void* d_ws, size_t ws_size,
                              hipStream_t stream) {
    const float* x         = (const float*)d_in[0];
    const float* c_attn_w  = (const float*)d_in[1];
    const float* c_attn_b  = (const float*)d_in[2];
    const float* c_proj_w  = (const float*)d_in[3];
    const float* c_proj_b  = (const float*)d_in[4];

    float* out     = (float*)d_out;
    float* present = out + AELEMS;           // [B,2,NH,S,HD] f32, written by qkv kernel

    __bf16* qws = (__bf16*)d_ws;             // Q bf16 [B,NH,S,HD]
    __bf16* aws = qws + AELEMS;              // merged attn out bf16 [B,S,NX]

    qkv_gemm_kernel<<<dim3((3 * NXX) / 64, MTOT / 128), 256, 0, stream>>>(
        x, c_attn_w, c_attn_b, present, qws);

    attn_flash_kernel<<<dim3(SS / 64, NHH, BQ), 128, 0, stream>>>(
        qws, present, aws);

    proj_gemm_kernel<<<dim3(NXX / 64, MTOT / 128), 256, 0, stream>>>(
        aws, c_proj_w, c_proj_b, out);
}
